// GRAMAttention_34119220199535
// MI455X (gfx1250) — matrix operand order repaired and checked by reading, not compiled, not run on hardware
//
#include <hip/hip_runtime.h>

// ---------------------------------------------------------------------------
// MI455X (gfx1250) attention pipeline, bf16 WMMA (f32 accum) throughout.
// Shapes: B=2, T=2048, D=1024, H=16, hd=64.  All buffers L2-resident.
// ---------------------------------------------------------------------------

typedef __attribute__((ext_vector_type(16))) __bf16 v16bf;
typedef __attribute__((ext_vector_type(8)))  float  v8f;
typedef __attribute__((ext_vector_type(4)))  int    v4i;
typedef unsigned short u16;

union FragB { v4i i[2]; v16bf v; };   // 16 bf16 per lane = one WMMA A/B operand

__device__ __forceinline__ u16 f2bf(float f) {
  unsigned u = __float_as_uint(f);
  u = (u + 0x7fffu + ((u >> 16) & 1u)) >> 16;   // round-to-nearest-even
  return (u16)u;
}
__device__ __forceinline__ v4i ldg128(const u16* p) { return *(const v4i*)p; }

// ---------------------------------------------------------------------------
// f32 -> bf16 elementwise convert
// ---------------------------------------------------------------------------
__global__ void __launch_bounds__(256)
cvt_bf16(const float* __restrict__ s, u16* __restrict__ d, int n) {
  int i = blockIdx.x * 256 + threadIdx.x;
  if (i < n) d[i] = f2bf(s[i]);
}

// ---------------------------------------------------------------------------
// GEMM: C(4096x1024) = A(4096x1024,bf16) @ W(1024x1024,bf16,N-major).T
// Block = 256 thr = 8 waves; block tile 128x128; wave tile 64x32 (4x2 WMMA).
// mode 0: RoPE + store Q -> (B,H,T,hd) bf16
// mode 1: RoPE + store K -> (B,H,T,hd) bf16
// mode 2: store V transposed -> (B,H,hd,T) bf16   (makes PV B-operand contiguous)
// mode 3: store f32 (final output projection)
// ---------------------------------------------------------------------------
__global__ void __launch_bounds__(256)
proj_gemm(const u16* __restrict__ A, const u16* __restrict__ W,
          const float* __restrict__ fc, u16* __restrict__ obf,
          float* __restrict__ of32, int mode)
{
  const int lane = threadIdx.x & 31;
  const int wib  = threadIdx.x >> 5;
  const int hi   = lane >> 4;         // lane-half selects K sub-chunks
  const int ln   = lane & 15;
  const int rowB = blockIdx.y * 128 + (wib & 1) * 64;
  const int colB = blockIdx.x * 128 + (wib >> 1) * 32;

  const v8f zero8 = {0.f,0.f,0.f,0.f,0.f,0.f,0.f,0.f};
  v8f acc[4][2];
#pragma unroll
  for (int ms = 0; ms < 4; ++ms) { acc[ms][0] = zero8; acc[ms][1] = zero8; }

  for (int k0 = 0; k0 < 1024; k0 += 32) {
    FragB af[4], bfr[2];
#pragma unroll
    for (int ms = 0; ms < 4; ++ms) {
      // A layout: lane<16 -> K {0..7,16..23}; lane>=16 -> {8..15,24..31}
      const u16* pa = A + (rowB + ms*16 + ln) * 1024 + k0 + hi*8;
      af[ms].i[0] = ldg128(pa);
      af[ms].i[1] = ldg128(pa + 16);
    }
#pragma unroll
    for (int ns = 0; ns < 2; ++ns) {
      // B layout: lane<16 -> K {0..7,16..23}; lane>=16 -> {8..15,24..31}
      const u16* pb = W + (colB + ns*16 + ln) * 1024 + k0 + hi*8;
      bfr[ns].i[0] = ldg128(pb);
      bfr[ns].i[1] = ldg128(pb + 16);
    }
#pragma unroll
    for (int ms = 0; ms < 4; ++ms)
#pragma unroll
      for (int ns = 0; ns < 2; ++ns)
        acc[ms][ns] = __builtin_amdgcn_wmma_f32_16x16x32_bf16(
            false, af[ms].v, false, bfr[ns].v, (short)0, acc[ms][ns],
            false, false);
  }

  // Epilogue. C layout: row = r + 8*(lane>=16), col = lane%16.
#pragma unroll
  for (int ms = 0; ms < 4; ++ms) {
#pragma unroll
    for (int ns = 0; ns < 2; ++ns) {
#pragma unroll
      for (int r = 0; r < 8; ++r) {
        const int m = rowB + ms*16 + r + 8*hi;   // token row 0..4095
        const int n = colB + ns*16 + ln;         // feature 0..1023
        float v = acc[ms][ns][r];
        if (mode == 3) {
          of32[m*1024 + n] = v;
        } else {
          const int b = m >> 11, t = m & 2047;
          const int h = n >> 6,  dh = n & 63;
          if (mode == 2) {
            obf[((b*16 + h)*64 + dh)*2048 + t] = f2bf(v);      // V transposed
          } else {
            // RoPE: pair partner lives in adjacent lane (n ^ 1)
            const float c = fc[(t*32 + (dh >> 1))*2 + 0];
            const float s = fc[(t*32 + (dh >> 1))*2 + 1];
            const float partner = __shfl_xor(v, 1, 32);
            const float o = (dh & 1) ? (partner*s + v*c) : (v*c - partner*s);
            obf[((b*16 + h)*2048 + t)*64 + dh] = f2bf(o);
          }
        }
      }
    }
  }
}

// ---------------------------------------------------------------------------
// Flash attention, causal.  One wave per 16-row q tile, 32 keys per step.
// Q/K in (B,H,T,64) bf16 (post-RoPE), V in (B,H,64,T) bf16 (transposed).
// ---------------------------------------------------------------------------
__global__ void __launch_bounds__(256)
attn_flash(const u16* __restrict__ Q, const u16* __restrict__ K,
           const u16* __restrict__ Vt, u16* __restrict__ aout)
{
  __shared__ __attribute__((aligned(16))) u16 Plds[8][16*32]; // P staging / wave

  const int lane = threadIdx.x & 31;
  const int wib  = threadIdx.x >> 5;
  const int hi = lane >> 4, ln = lane & 15;
  const int w  = blockIdx.x * 8 + wib;         // 4096 waves total
  const int b  = w >> 11;
  const int h  = (w >> 7) & 15;
  const int qb = (w & 127) * 16;

  const u16* Qh = Q  + (b*16 + h) * 2048 * 64;
  const u16* Kh = K  + (b*16 + h) * 2048 * 64;
  const u16* Vh = Vt + (b*16 + h) * 64 * 2048;
  u16* Pl = &Plds[wib][0];

  // Q A-operands for k-dims [0,32) and [32,64)
  FragB aq[2];
#pragma unroll
  for (int s = 0; s < 2; ++s) {
    const u16* p = Qh + (qb + ln)*64 + s*32 + hi*8;
    aq[s].i[0] = ldg128(p);
    aq[s].i[1] = ldg128(p + 16);
  }

  const v8f zero8 = {0.f,0.f,0.f,0.f,0.f,0.f,0.f,0.f};
  v8f o[4] = {zero8, zero8, zero8, zero8};     // 16x64 f32 output accum
  float mrow[8], lrow[8];
#pragma unroll
  for (int r = 0; r < 8; ++r) { mrow[r] = -1e30f; lrow[r] = 0.f; }

  const float scale = 0.125f;                  // 1/sqrt(64)

  for (int kb = 0; kb < qb + 16; kb += 32) {
    // ---- scores: S(16x32) = Q(16x64) x K^T(64x32), two 16x16 tiles ----
    v8f sc[2];
#pragma unroll
    for (int nt = 0; nt < 2; ++nt) {
      sc[nt] = zero8;
#pragma unroll
      for (int ks = 0; ks < 2; ++ks) {
        FragB bk;   // column = key index, K-dim contiguous in Kh row
        const u16* p = Kh + (kb + nt*16 + ln)*64 + ks*32 + hi*8;
        bk.i[0] = ldg128(p); bk.i[1] = ldg128(p + 16);
        sc[nt] = __builtin_amdgcn_wmma_f32_16x16x32_bf16(
            false, aq[ks].v, false, bk.v, (short)0, sc[nt], false, false);
      }
    }
    // ---- scale + causal mask (analytic) ----
#pragma unroll
    for (int nt = 0; nt < 2; ++nt)
#pragma unroll
      for (int r = 0; r < 8; ++r) {
        const int key  = kb + nt*16 + ln;
        const int qrow = qb + r + 8*hi;
        sc[nt][r] = sc[nt][r]*scale + (key <= qrow ? 0.f : -1e30f);
      }
    // ---- online softmax: rows per-VGPR, cols per-lane -> 16-lane reductions
    float mnew[8], alpha[8];
#pragma unroll
    for (int r = 0; r < 8; ++r) {
      float v = fmaxf(sc[0][r], sc[1][r]);
#pragma unroll
      for (int msk = 1; msk < 16; msk <<= 1) v = fmaxf(v, __shfl_xor(v, msk, 32));
      mnew[r] = fmaxf(mrow[r], v);
    }
#pragma unroll
    for (int nt = 0; nt < 2; ++nt)
#pragma unroll
      for (int r = 0; r < 8; ++r)
        sc[nt][r] = __expf(sc[nt][r] - mnew[r]);
#pragma unroll
    for (int r = 0; r < 8; ++r) {
      float v = sc[0][r] + sc[1][r];
#pragma unroll
      for (int msk = 1; msk < 16; msk <<= 1) v += __shfl_xor(v, msk, 32);
      alpha[r] = __expf(mrow[r] - mnew[r]);
      lrow[r]  = lrow[r]*alpha[r] + v;
      mrow[r]  = mnew[r];
    }
#pragma unroll
    for (int j = 0; j < 4; ++j)
#pragma unroll
      for (int r = 0; r < 8; ++r)
        o[j][r] *= alpha[r];

    // ---- P: C-layout -> A-layout via per-wave LDS round trip ----
#pragma unroll
    for (int nt = 0; nt < 2; ++nt)
#pragma unroll
      for (int r = 0; r < 8; ++r)
        Pl[(r + 8*hi)*32 + nt*16 + ln] = f2bf(sc[nt][r]);
    asm volatile("s_wait_dscnt 0x0" ::: "memory");  // wave-local LDS ordering
    FragB pf;
    {
      const u16* pp = Pl + ln*32 + hi*8;
      pf.i[0] = *(const v4i*)pp;
      pf.i[1] = *(const v4i*)(pp + 16);
    }

    // ---- O(16x64) += P(16x32) x V(32x64); V transposed -> contiguous B ----
#pragma unroll
    for (int j = 0; j < 4; ++j) {
      FragB bv;
      const u16* vp = Vh + (j*16 + ln)*2048 + kb + hi*8;
      bv.i[0] = ldg128(vp); bv.i[1] = ldg128(vp + 16);
      o[j] = __builtin_amdgcn_wmma_f32_16x16x32_bf16(
          false, pf.v, false, bv.v, (short)0, o[j], false, false);
    }
  }

  // ---- normalize + store (B,T,D) bf16 ----
#pragma unroll
  for (int r = 0; r < 8; ++r) lrow[r] = 1.f / lrow[r];
#pragma unroll
  for (int j = 0; j < 4; ++j)
#pragma unroll
    for (int r = 0; r < 8; ++r) {
      const int t   = qb + r + 8*hi;
      const int col = h*64 + j*16 + ln;
      aout[(b*2048 + t)*1024 + col] = f2bf(o[j][r] * lrow[r]);
    }
}

// ---------------------------------------------------------------------------
// Launch: cvt x5 -> QKV proj (RoPE / V-transpose fused) -> flash attn -> O proj
// Workspace layout (bytes): Xbf 8M | Wq 2M | Wk 2M | Wv 2M | Wo 2M |
//                           Qr 8M | Kr 8M | Vt 8M | AO 8M   (total 48 MiB)
// ---------------------------------------------------------------------------
extern "C" void kernel_launch(void* const* d_in, const int* in_sizes, int n_in,
                              void* d_out, int out_size, void* d_ws, size_t ws_size,
                              hipStream_t stream)
{
  const float* x  = (const float*)d_in[0];
  const float* wq = (const float*)d_in[1];
  const float* wk = (const float*)d_in[2];
  const float* wv = (const float*)d_in[3];
  const float* wo = (const float*)d_in[4];
  const float* fc = (const float*)d_in[5];
  // d_in[6] (causal_mask) is applied analytically in attn_flash.

  char* ws = (char*)d_ws;
  const size_t MB = 1u << 20;
  u16* Xbf = (u16*)(ws);
  u16* Wqb = (u16*)(ws +  8*MB);
  u16* Wkb = (u16*)(ws + 10*MB);
  u16* Wvb = (u16*)(ws + 12*MB);
  u16* Wob = (u16*)(ws + 14*MB);
  u16* Qr  = (u16*)(ws + 16*MB);
  u16* Kr  = (u16*)(ws + 24*MB);
  u16* Vt  = (u16*)(ws + 32*MB);
  u16* AO  = (u16*)(ws + 40*MB);

  const int nx = 2*2048*1024, nw = 1024*1024;
  cvt_bf16<<<(nx + 255)/256, 256, 0, stream>>>(x,  Xbf, nx);
  cvt_bf16<<<(nw + 255)/256, 256, 0, stream>>>(wq, Wqb, nw);
  cvt_bf16<<<(nw + 255)/256, 256, 0, stream>>>(wk, Wkb, nw);
  cvt_bf16<<<(nw + 255)/256, 256, 0, stream>>>(wv, Wvb, nw);
  cvt_bf16<<<(nw + 255)/256, 256, 0, stream>>>(wo, Wob, nw);

  dim3 g(8, 32);   // 1024/128 cols x 4096/128 rows
  proj_gemm<<<g, 256, 0, stream>>>(Xbf, Wqb, fc, Qr, nullptr, 0);
  proj_gemm<<<g, 256, 0, stream>>>(Xbf, Wkb, fc, Kr, nullptr, 1);
  proj_gemm<<<g, 256, 0, stream>>>(Xbf, Wvb, fc, Vt, nullptr, 2);

  attn_flash<<<512, 256, 0, stream>>>(Qr, Kr, Vt, AO);

  proj_gemm<<<g, 256, 0, stream>>>(AO, Wob, fc, nullptr, (float*)d_out, 3);
}
